// BinarizeConv2d_6983616824293
// MI455X (gfx1250) — matrix operand order repaired and checked by reading, not compiled, not run on hardware
//
#include <hip/hip_runtime.h>

// ---------------------------------------------------------------------------
// BinarizeConv2d on gfx1250 (MI455X): implicit GEMM with v_wmma_f32_16x16x32_f16
//
//   out[b][co][h][w] = sum_{cin,kh,kw} clip(x)[b][cin][h+kh-1][w+kw-1] * Wc[co][cin][kh][kw]
//   Wc rows 0..123 = clip(reweight), rows 124..127 = conv_w (concat branch)
//
// GEMM view: D[16 spatial][16 cout] += A[16 spatial][32 K] * B[32 K][16 cout]
// K ordering: ks = (kh*4 + ci)*3 + kw, klocal = cin within 32-tile  (36 steps)
// One 114-wide padded activation row staged per (kh,ci); the three kw taps
// reuse it at row offset +kw. Double-buffered LDS + 2-deep register pipeline:
// one barrier per staging step; fetch/commit both hide under the 21 WMMAs.
// ---------------------------------------------------------------------------

typedef __attribute__((ext_vector_type(16))) _Float16 v16h;
typedef __attribute__((ext_vector_type(8)))  float    v8f;

#define HW     112
#define CIN    128
#define COUT   128
#define BATCH  32
#define HW2    ((size_t)HW * HW)

// ---------------------------------------------------------------------------
// Weight packing: f16 fragments in exact per-lane WMMA B-matrix order.
// Layout: wf[ks][ct(8)][lane(32)][f(16)] halves. 36*8*32*16 = 147456 halves.
// B 32x16 f16 packing used here (contiguous, same as A): lane l -> column n = l%16; K-local = (l&16) + f.
// ---------------------------------------------------------------------------
__global__ __launch_bounds__(256) void pack_w_kernel(
    const float* __restrict__ rw,     // [124,128,3,3]
    const float* __restrict__ cwp,    // [4,128,3,3]
    _Float16* __restrict__ wf) {
  int idx  = blockIdx.x * 256 + threadIdx.x;    // < 147456, exact grid
  int f    = idx & 15;
  int l    = (idx >> 4) & 31;
  int ct   = (idx >> 9) & 7;
  int ks   = idx >> 12;                          // 0..35
  int cout = ct * 16 + (l & 15);
  int klocal = (l & 16) + f;                     // 0..31
  int kw   = ks % 3;
  int t    = ks / 3;                             // kh*4 + ci
  int ci   = t & 3;
  int kh   = t >> 2;
  int cin  = ci * 32 + klocal;
  float val;
  if (cout < 124) {
    val = rw[(((size_t)cout * CIN + cin) * 3 + kh) * 3 + kw];
    val = __builtin_amdgcn_fmed3f(val, -1.0f, 1.0f);   // binarized-weight clip
  } else {
    val = cwp[(((size_t)(cout - 124) * CIN + cin) * 3 + kh) * 3 + kw];
  }
  wf[idx] = (_Float16)val;
}

// ---------------------------------------------------------------------------
// Main conv kernel. Block = 256 threads = 8 waves. Tile: 112 spatial x 128 cout.
// ---------------------------------------------------------------------------
struct alignas(16) Smem {
  union {
    _Float16 a[2][114][40];  // double-buffered activation row [buf][c][k], 80 B stride
    float    o[128][116];    // epilogue transpose buffer [cout][n]
  };
};

__global__ __launch_bounds__(256) void conv_wmma_kernel(
    const float* __restrict__ x,      // [32,128,112,112] f32
    const _Float16* __restrict__ wf,  // packed weight fragments
    float* __restrict__ out) {        // [32,128,112,112] f32
  __shared__ Smem sm;

  const int tid  = threadIdx.x;
  const int lane = tid & 31;
  const int wave = tid >> 5;          // cout tile 0..7

  const int bidx = blockIdx.x;        // grid = 32*112 = 3584, one per (b,h)
  const int h    = bidx % HW;
  const int bb   = bidx / HW;

  const int nrow = lane & 15;         // A row / D column index
  const int hi   = lane >> 4;         // lane half
  const int b0   = hi * 8;            // M base for D rows

  // staging mapping: column c holds x at w = c-1 (c=0,113 are conv padding)
  const int sc = tid & 127;           // column 0..127 (active if < 114)
  const int k0 = (tid >> 7) * 16;     // k base: 0 or 16
  const bool cvalid = (sc >= 1) & (sc <= HW);
  const int wwc = cvalid ? (sc - 1) : 0;   // clamped, always in-bounds

  const float* xb = x + (size_t)bb * CIN * HW2;

  // two-deep register pipeline of raw f32 staged values
  float stg0[16], stg1[16];
  auto stage_fetch = [&](int t, float* s) {
    const int kh = t >> 2;            // 0..2
    const int ci = t & 3;             // cin tile
    const int hh = h + kh - 1;
    const bool valid = cvalid & (hh >= 0) & (hh < HW);
    const int hhc = (hh < 0) ? 0 : ((hh > HW - 1) ? HW - 1 : hh);
    const float* src = xb + ((size_t)(ci * 32 + k0) * HW + hhc) * HW + wwc;
    #pragma unroll
    for (int j = 0; j < 16; ++j)
      s[j] = valid ? src[(size_t)j * HW2] : 0.0f;   // coalesced across lanes
  };
  auto commit = [&](int t, const float* s) {        // clip+cvt, 2x ds_store_b128
    if (sc < 114) {
      alignas(16) _Float16 tmp[16];
      #pragma unroll
      for (int j = 0; j < 16; ++j)
        tmp[j] = (_Float16)__builtin_amdgcn_fmed3f(s[j], -1.0f, 1.0f);
      _Float16* dst = &sm.a[t & 1][sc][k0];
      *reinterpret_cast<uint4*>(dst)     = *reinterpret_cast<const uint4*>(tmp);
      *reinterpret_cast<uint4*>(dst + 8) = *reinterpret_cast<const uint4*>(tmp + 8);
    }
  };

  v8f acc[7];
  #pragma unroll
  for (int i = 0; i < 7; ++i)
    #pragma unroll
    for (int j = 0; j < 8; ++j) acc[i][j] = 0.0f;

  const uint4* wbase = reinterpret_cast<const uint4*>(wf) +
                       ((size_t)wave * 32 + lane) * 2;   // + ks*512 per step

  // prologue: fill buffer 0, issue fetch for step 1
  stage_fetch(0, stg0);
  stage_fetch(1, stg1);
  commit(0, stg0);

  for (int t = 0; t < 12; ++t) {      // t = kh*4 + ci
    __syncthreads();                  // commit(t) visible; buf[(t+1)&1] readers done

    // commit(t+1): consumes loads issued one full compute-phase ago (no stall),
    // writes the opposite-parity buffer from this step's reads.
    if (t + 1 < 12) commit(t + 1, ((t + 1) & 1) ? stg1 : stg0);
    // fetch(t+2): global loads hide under the 21 WMMAs below.
    if (t + 2 < 12) stage_fetch(t + 2, (t & 1) ? stg1 : stg0);
    if (t + 1 < 12)
      __builtin_prefetch(reinterpret_cast<const void*>(wbase + (size_t)(t + 1) * 3 * 512),
                         0, 0);       // -> global_prefetch_b8 (next weights)

    // ---- three kw taps reuse the staged row at offset +kw ----
    const _Float16(*abuf)[40] = sm.a[t & 1];
    #pragma unroll
    for (int kw = 0; kw < 3; ++kw) {
      const int ks = t * 3 + kw;
      const uint4* wp = wbase + (size_t)ks * 512;
      union { uint4 u[2]; v16h v; } bw;
      bw.u[0] = wp[0];
      bw.u[1] = wp[1];
      #pragma unroll
      for (int mt = 0; mt < 7; ++mt) {
        const uint4* rp = reinterpret_cast<const uint4*>(&abuf[mt * 16 + nrow + kw][0]);
        union { uint4 u[2]; v16h v; } aw;
        aw.u[0] = rp[2 * hi];         // K = 2*b0 .. 2*b0+7
        aw.u[1] = rp[2 * hi + 1];     // K = 2*b0+8 .. 2*b0+15
        acc[mt] = __builtin_amdgcn_wmma_f32_16x16x32_f16(
            /*neg_a=*/false, aw.v, /*neg_b=*/false, bw.v,
            /*c_mod=*/(short)0, acc[mt], /*reuse_a=*/false, /*reuse_b=*/false);
      }
    }
  }

  // ---- epilogue: transpose through LDS -> coalesced float4 stores ----
  __syncthreads();
  #pragma unroll
  for (int mt = 0; mt < 7; ++mt)
    #pragma unroll
    for (int r = 0; r < 8; ++r)
      sm.o[wave * 16 + nrow][mt * 16 + r + b0] = acc[mt][r];
  __syncthreads();

  float* ob = out + ((size_t)bb * COUT * HW + h) * HW;
  #pragma unroll
  for (int it = 0; it < 14; ++it) {
    int flat = it * 256 + tid;        // 0..3583 -> (cout, w/4); 128*28 exact
    int cout = flat / 28;
    int n    = (flat - cout * 28) * 4;
    float4 val = *reinterpret_cast<const float4*>(&sm.o[cout][n]);
    *reinterpret_cast<float4*>(ob + (size_t)cout * HW2 + n) = val;
  }
}

// ---------------------------------------------------------------------------
extern "C" void kernel_launch(void* const* d_in, const int* in_sizes, int n_in,
                              void* d_out, int out_size, void* d_ws, size_t ws_size,
                              hipStream_t stream) {
  (void)in_sizes; (void)n_in; (void)out_size; (void)ws_size;
  const float* x   = (const float*)d_in[0];   // [32,128,112,112]
  const float* rw  = (const float*)d_in[1];   // [124,128,3,3]
  const float* cwp = (const float*)d_in[2];   // [4,128,3,3]
  float* out = (float*)d_out;
  _Float16* wf = (_Float16*)d_ws;             // 147456 halves = 294912 B

  // 1) pack weights into per-lane WMMA fragment order (tiny, L2-resident)
  pack_w_kernel<<<576, 256, 0, stream>>>(rw, cwp, wf);
  // 2) implicit-GEMM conv: one block per (batch, h) row
  conv_wmma_kernel<<<BATCH * HW, 256, 0, stream>>>(x, wf, out);
}
